// DeformableMNIST_15290083574327
// MI455X (gfx1250) — matrix operand order repaired and checked by reading, not compiled, not run on hardware
//
#include <hip/hip_runtime.h>

typedef __attribute__((ext_vector_type(16))) _Float16 v16h;
typedef __attribute__((ext_vector_type(8)))  _Float16 v8h;
typedef __attribute__((ext_vector_type(2)))  _Float16 v2h;
typedef __attribute__((ext_vector_type(8)))  float    v8f;

namespace {
constexpr int BATCH = 1024;
constexpr int P28   = 784;    // 28*28
constexpr int P14   = 196;    // 14*14
constexpr int C1    = 32;
constexpr int C2    = 64;
constexpr int KD2   = 288;    // C1 * 9
constexpr long KFC  = 12544;  // C2 * 196
constexpr int NFC   = 128;
constexpr int LDSA  = 290;    // 288 + 2 pad -> 145-dword row stride, conflict-free
}

// zero-padded bilinear sample (matches reference: clip + per-corner valid mask)
__device__ __forceinline__ float bilin(const float* img, int HH, int WW, float py, float px) {
    float y0f = floorf(py), x0f = floorf(px);
    int y0 = (int)y0f, x0 = (int)x0f;
    float wy1 = py - y0f, wx1 = px - x0f;
    float wy0 = 1.f - wy1, wx0 = 1.f - wx1;
    float v00 = (y0   >= 0 && y0   < HH && x0   >= 0 && x0   < WW) ? img[y0*WW + x0]       : 0.f;
    float v01 = (y0   >= 0 && y0   < HH && x0+1 >= 0 && x0+1 < WW) ? img[y0*WW + x0+1]     : 0.f;
    float v10 = (y0+1 >= 0 && y0+1 < HH && x0   >= 0 && x0   < WW) ? img[(y0+1)*WW + x0]   : 0.f;
    float v11 = (y0+1 >= 0 && y0+1 < HH && x0+1 >= 0 && x0+1 < WW) ? img[(y0+1)*WW + x0+1] : 0.f;
    return v00*wy0*wx0 + v01*wy0*wx1 + v10*wy1*wx0 + v11*wy1*wx1;
}

// A-fragment loader: row-major 16-bit A, 16x32 tile in contiguous k order (same order as load_b_frag)
// element pair j lives at K = half*16 + j*2
__device__ __forceinline__ v16h load_a_frag(const _Float16* Ar, int half) {
    v16h a;
    #pragma unroll
    for (int j = 0; j < 8; ++j) {
        int kp = (half << 4) + (j << 1);
        v2h pr = *(const v2h*)(Ar + kp);
        a[2*j] = pr[0]; a[2*j+1] = pr[1];
    }
    return a;
}

// B-fragment loader: (N,K) row-major 16-bit weights; lane n = lane&15, K = half*16 + i
__device__ __forceinline__ v16h load_b_frag(const _Float16* Br) {
    v16h bb;
    v8h lo = *(const v8h*)Br;
    v8h hi = *(const v8h*)(Br + 8);
    #pragma unroll
    for (int i = 0; i < 8; ++i) { bb[i] = lo[i]; bb[8+i] = hi[i]; }
    return bb;
}

// ---------------- f32 -> f16 copy ----------------
__global__ __launch_bounds__(256) void k_cvt_f16(const float* __restrict__ src,
                                                 _Float16* __restrict__ dst, int n) {
    int i = blockIdx.x * 256 + threadIdx.x;
    if (i < n) dst[i] = (_Float16)src[i];
}

// ---------- stage 1: conv1(offsets) + deform1 + relu + 2x2 pool, Cin=1 ----------
// one thread per (b, 14x14 pixel); writes h1 (B, 32, 14, 14) f32
__global__ __launch_bounds__(256) void k_stage1(const float* __restrict__ x,
                                                const float* __restrict__ ow1,
                                                const float* __restrict__ ob1,
                                                const float* __restrict__ w1,
                                                const float* __restrict__ b1,
                                                float* __restrict__ h1) {
    int id = blockIdx.x * 256 + threadIdx.x;            // 200704 total, exact
    int b = id / P14, p = id % P14;
    int y14 = p / 14, x14 = p % 14;
    const float* img = x + (size_t)b * P28;

    float pool[C1];
    #pragma unroll
    for (int c = 0; c < C1; ++c) pool[c] = 0.f;

    #pragma unroll
    for (int sy = 0; sy < 2; ++sy)
    #pragma unroll
    for (int sx = 0; sx < 2; ++sx) {
        int y = 2*y14 + sy, xx = 2*x14 + sx;
        float xs[9];
        #pragma unroll
        for (int i = 0; i < 3; ++i)
        #pragma unroll
        for (int j = 0; j < 3; ++j) {
            int yy = y - 1 + i, xc = xx - 1 + j;
            xs[i*3+j] = (yy >= 0 && yy < 28 && xc >= 0 && xc < 28) ? img[yy*28 + xc] : 0.f;
        }
        float off[18];
        #pragma unroll
        for (int o = 0; o < 18; ++o) {
            float v = ob1[o];
            #pragma unroll
            for (int t = 0; t < 9; ++t) v += xs[t] * ow1[o*9 + t];
            off[o] = v;
        }
        float s[9];
        #pragma unroll
        for (int k = 0; k < 9; ++k) {
            float py = off[2*k]   + (float)(y  - 1) + (float)(k / 3);
            float px = off[2*k+1] + (float)(xx - 1) + (float)(k % 3);
            s[k] = bilin(img, 28, 28, py, px);
        }
        #pragma unroll
        for (int c = 0; c < C1; ++c) {
            float v = b1[c];
            #pragma unroll
            for (int k = 0; k < 9; ++k) v += s[k] * w1[c*9 + k];
            pool[c] += fmaxf(v, 0.f) * 0.25f;
        }
    }
    #pragma unroll
    for (int c = 0; c < C1; ++c)
        h1[((size_t)b * C1 + c) * P14 + p] = pool[c];
}

// ---------- stage 2a: conv2 (32 -> 18) on 14x14, writes offset2 (B,18,14,14) ----------
__global__ __launch_bounds__(256) void k_conv2(const float* __restrict__ h1,
                                               const float* __restrict__ ow2,
                                               const float* __restrict__ ob2,
                                               float* __restrict__ off2) {
    int id = blockIdx.x * 256 + threadIdx.x;            // 200704, exact
    int b = id / P14, p = id % P14;
    int y = p / 14, x = p % 14;
    float acc[18];
    #pragma unroll
    for (int o = 0; o < 18; ++o) acc[o] = ob2[o];
    for (int c = 0; c < C1; ++c) {
        const float* img = h1 + ((size_t)b * C1 + c) * P14;
        float v[9];
        #pragma unroll
        for (int i = 0; i < 3; ++i)
        #pragma unroll
        for (int j = 0; j < 3; ++j) {
            int yy = y - 1 + i, xc = x - 1 + j;
            v[i*3+j] = (yy >= 0 && yy < 14 && xc >= 0 && xc < 14) ? img[yy*14 + xc] : 0.f;
        }
        #pragma unroll
        for (int o = 0; o < 18; ++o) {
            const float* wv = ow2 + ((size_t)o * C1 + c) * 9;
            float a = acc[o];
            #pragma unroll
            for (int t = 0; t < 9; ++t) a += v[t] * wv[t];
            acc[o] = a;
        }
    }
    #pragma unroll
    for (int o = 0; o < 18; ++o)
        off2[((size_t)b * 18 + o) * P14 + p] = acc[o];
}

// ---------- stage 2b: fused deformable im2col (LDS) + WMMA GEMM ----------
// block = 128 threads = 4 waves; block handles 16 rows (M) x all 64 cols (N).
// M = B*196 = 200704 rows -> 12544 blocks. K = 288, 9 wmma steps.
// Output: relu(out + b2) -> f16 into A3 laid out as flatten of (B,64,14,14).
__global__ __launch_bounds__(128) void k_dgemm2(const float* __restrict__ h1,
                                                const float* __restrict__ off2,
                                                const _Float16* __restrict__ w2h,
                                                const float* __restrict__ b2,
                                                _Float16* __restrict__ A3) {
    __shared__ float s_py[16*9];
    __shared__ float s_px[16*9];
    __shared__ _Float16 sA[16 * LDSA];

    const int tid  = threadIdx.x;
    const int lane = tid & 31;
    const int wave = tid >> 5;
    const int m0   = blockIdx.x << 4;

    // phase 0: sampling positions for 16 rows x 9 taps
    for (int idx = tid; idx < 144; idx += 128) {
        int r = idx / 9, k = idx % 9;
        int m = m0 + r, b = m / P14, p = m % P14;
        int y = p / 14, x = p % 14;
        float dy = off2[((size_t)b * 18 + 2*k    ) * P14 + p];
        float dx = off2[((size_t)b * 18 + 2*k + 1) * P14 + p];
        s_py[idx] = dy + (float)(y - 1) + (float)(k / 3);
        s_px[idx] = dx + (float)(x - 1) + (float)(k % 3);
    }
    __syncthreads();

    // phase 1: bilinear-sample 16x288 A tile into LDS (f16)
    for (int idx = tid; idx < 16 * KD2; idx += 128) {
        int r = idx / KD2, kk = idx % KD2;
        int c = kk / 9, k = kk % 9;
        int b = (m0 + r) / P14;
        float v = bilin(h1 + ((size_t)b * C1 + c) * P14, 14, 14,
                        s_py[r*9 + k], s_px[r*9 + k]);
        sA[r * LDSA + kk] = (_Float16)v;
    }
    __syncthreads();

    // phase 2: WMMA — wave w covers columns n = w*16 .. w*16+15
    const int half = lane >> 4;
    const int mr   = lane & 15;
    const int n    = (wave << 4) + (lane & 15);
    v8f acc = {};
    #pragma unroll
    for (int t = 0; t < 9; ++t) {
        const int kb = t * 32;
        v16h a  = load_a_frag(&sA[mr * LDSA + kb], half);
        v16h bb = load_b_frag(w2h + (size_t)n * KD2 + kb + (half << 4));
        acc = __builtin_amdgcn_wmma_f32_16x16x32_f16(false, a, false, bb,
                                                     (short)0, acc, false, false);
    }
    const float bias = b2[n];
    #pragma unroll
    for (int r = 0; r < 8; ++r) {
        int m = m0 + r + (half << 3);
        int b = m / P14, p = m % P14;
        float v = fmaxf(acc[r] + bias, 0.f);   // pool2 is identity (14//14==1)
        A3[(size_t)b * KFC + (size_t)n * P14 + p] = (_Float16)v;
    }
}

// ---------- stage 3: fc1 WMMA GEMM: (1024 x 12544) x (12544 x 128) ----------
// Each wave owns one 16-row M tile and FOUR 16-col N tiles (64 cols): the A
// fragment is loaded once per K-step and reused by 4 wmma ops (4x less A
// traffic, 4x higher wmma density). 64 M-tiles x 2 N-halves = 128 waves
// -> 32 blocks of 128 threads. B rows are streamed once; prefetch ahead.
__global__ __launch_bounds__(128) void k_fc1(const _Float16* __restrict__ A3,
                                             const _Float16* __restrict__ fw1h,
                                             const float* __restrict__ fb1,
                                             float* __restrict__ fc1o) {
    const int tid   = threadIdx.x;
    const int lane  = tid & 31;
    const int wid   = blockIdx.x * 4 + (tid >> 5);   // 0..127
    const int m0    = (wid >> 1) << 4;               // 64 M-tiles
    const int nbase = (wid & 1) << 6;                // 0 or 64
    const int half  = lane >> 4;
    const int mr    = lane & 15;
    const int nl    = lane & 15;

    const _Float16* Arow = A3 + (size_t)(m0 + mr) * KFC;
    const _Float16* Br0  = fw1h + (size_t)(nbase      + nl) * KFC + (half << 4);
    const _Float16* Br1  = fw1h + (size_t)(nbase + 16 + nl) * KFC + (half << 4);
    const _Float16* Br2  = fw1h + (size_t)(nbase + 32 + nl) * KFC + (half << 4);
    const _Float16* Br3  = fw1h + (size_t)(nbase + 48 + nl) * KFC + (half << 4);

    v8f acc0 = {}, acc1 = {}, acc2 = {}, acc3 = {};
    for (int kb = 0; kb < (int)KFC; kb += 32) {
        if (kb + 512 < (int)KFC) {              // prefetch B stream 16 steps ahead
            __builtin_prefetch(Br0 + kb + 512, 0, 1);
            __builtin_prefetch(Br1 + kb + 512, 0, 1);
            __builtin_prefetch(Br2 + kb + 512, 0, 1);
            __builtin_prefetch(Br3 + kb + 512, 0, 1);
        }
        v16h a = load_a_frag(Arow + kb, half);
        v16h b0 = load_b_frag(Br0 + kb);
        acc0 = __builtin_amdgcn_wmma_f32_16x16x32_f16(false, a, false, b0,
                                                      (short)0, acc0, false, false);
        v16h b1 = load_b_frag(Br1 + kb);
        acc1 = __builtin_amdgcn_wmma_f32_16x16x32_f16(false, a, false, b1,
                                                      (short)0, acc1, false, false);
        v16h b2 = load_b_frag(Br2 + kb);
        acc2 = __builtin_amdgcn_wmma_f32_16x16x32_f16(false, a, false, b2,
                                                      (short)0, acc2, false, false);
        v16h b3 = load_b_frag(Br3 + kb);
        acc3 = __builtin_amdgcn_wmma_f32_16x16x32_f16(false, a, false, b3,
                                                      (short)0, acc3, false, false);
    }
    #pragma unroll
    for (int sub = 0; sub < 4; ++sub) {
        const v8f& acc = (sub == 0) ? acc0 : (sub == 1) ? acc1 : (sub == 2) ? acc2 : acc3;
        const int n = nbase + sub * 16 + nl;
        const float bias = fb1[n];
        #pragma unroll
        for (int r = 0; r < 8; ++r) {
            int m = m0 + r + (half << 3);
            fc1o[(size_t)m * NFC + n] = fmaxf(acc[r] + bias, 0.f);  // relu for fc2
        }
    }
}

// ---------- stage 4: fc2 (128 -> 10), VALU ----------
__global__ __launch_bounds__(256) void k_fc2(const float* __restrict__ fc1o,
                                             const float* __restrict__ fw2,
                                             const float* __restrict__ fb2,
                                             float* __restrict__ out) {
    int id = blockIdx.x * 256 + threadIdx.x;   // 10240, exact
    int b = id / 10, o = id % 10;
    float v = fb2[o];
    const float* a = fc1o + (size_t)b * NFC;
    const float* w = fw2 + (size_t)o * NFC;
    #pragma unroll 8
    for (int j = 0; j < NFC; ++j) v += a[j] * w[j];
    out[id] = v;
}

extern "C" void kernel_launch(void* const* d_in, const int* in_sizes, int n_in,
                              void* d_out, int out_size, void* d_ws, size_t ws_size,
                              hipStream_t stream) {
    const float* x   = (const float*)d_in[0];
    const float* ow1 = (const float*)d_in[1];
    const float* ob1 = (const float*)d_in[2];
    const float* w1  = (const float*)d_in[3];
    const float* b1  = (const float*)d_in[4];
    const float* ow2 = (const float*)d_in[5];
    const float* ob2 = (const float*)d_in[6];
    const float* w2  = (const float*)d_in[7];
    const float* b2  = (const float*)d_in[8];
    const float* fw1 = (const float*)d_in[9];
    const float* fb1 = (const float*)d_in[10];
    const float* fw2 = (const float*)d_in[11];
    const float* fb2 = (const float*)d_in[12];
    float* out = (float*)d_out;

    // workspace carve-out (256B aligned slices), ~70 MB total
    char* wp = (char*)d_ws;
    auto carve = [&](size_t bytes) {
        void* p = (void*)wp;
        wp += (bytes + 255) & ~(size_t)255;
        return p;
    };
    float*    h1    = (float*)   carve((size_t)BATCH * C1 * P14 * 4);  // 25.7 MB
    float*    off2  = (float*)   carve((size_t)BATCH * 18 * P14 * 4);  // 14.5 MB
    _Float16* A3    = (_Float16*)carve((size_t)BATCH * KFC * 2);       // 25.7 MB
    _Float16* fw1h  = (_Float16*)carve((size_t)NFC * KFC * 2);         //  3.2 MB
    _Float16* w2h   = (_Float16*)carve((size_t)C2 * KD2 * 2);
    float*    fc1o  = (float*)   carve((size_t)BATCH * NFC * 4);
    (void)ws_size; (void)n_in; (void)in_sizes; (void)out_size;

    // weight conversions to f16 (row-major (N,K) preserved)
    {
        int n = C2 * KD2;                                   // 18432
        k_cvt_f16<<<(n + 255) / 256, 256, 0, stream>>>(w2, w2h, n);
    }
    {
        int n = NFC * (int)KFC;                             // 1605632
        k_cvt_f16<<<(n + 255) / 256, 256, 0, stream>>>(fw1, fw1h, n);
    }

    // stage 1: fused conv1 + deform1 + relu + 2x2 pool
    k_stage1<<<(BATCH * P14) / 256, 256, 0, stream>>>(x, ow1, ob1, w1, b1, h1);

    // stage 2a: conv2 -> offset2
    k_conv2<<<(BATCH * P14) / 256, 256, 0, stream>>>(h1, ow2, ob2, off2);

    // stage 2b: fused deformable im2col (LDS) + WMMA GEMM -> flattened f16 fc input
    k_dgemm2<<<(BATCH * P14) / 16, 128, 0, stream>>>(h1, off2, w2h, b2, A3);

    // stage 3: fc1 WMMA GEMM + bias + relu (4 N-tiles per wave, A reuse)
    k_fc1<<<32, 128, 0, stream>>>(A3, fw1h, fb1, fc1o);

    // stage 4: fc2
    k_fc2<<<(BATCH * 10) / 256, 256, 0, stream>>>(fc1o, fw2, fb2, out);
}
